// AttentionAgg2_76312978915994
// MI455X (gfx1250) — matrix operand order repaired and checked by reading, not compiled, not run on hardware
//
#include <hip/hip_runtime.h>
#include <hip/hip_bf16.h>

#define BB 32
#define SS 1024
#define EE 1024
#define NEGV -1e9f

typedef __bf16 bf16;
typedef __attribute__((ext_vector_type(16))) __bf16 v16bf;
typedef __attribute__((ext_vector_type(8)))  __bf16 v8bf;
typedef __attribute__((ext_vector_type(8)))  float  v8f;

__device__ inline v8f v8f_zero() {
  v8f z;
#pragma unroll
  for (int i = 0; i < 8; ++i) z[i] = 0.0f;
  return z;
}

__device__ inline v8f wmma_bf16(v16bf a, v16bf b, v8f c) {
  // D = A(16x32) * B(32x16) + C(16x16), f32 accumulate
  return __builtin_amdgcn_wmma_f32_16x16x32_bf16(
      /*neg_a=*/false, a, /*neg_b=*/false, b,
      /*c_mod=*/(short)0, c, /*reuse_a=*/false, /*reuse_b=*/false);
}

// A fragment (16x32, bf16, row-major source, leading dim ld):
// lane m = lane&15; grp = lane>>4.
// elems 0..7  = K = k + grp*16 .. +7
// elems 8..15 = K = k + grp*16 + 8 .. +15
__device__ inline v16bf load_a_frag(const bf16* base, int row, int k, int ld,
                                    int lm, int grp) {
  const bf16* p = base + (size_t)(row + lm) * ld + k + grp * 16;
  v8bf c0 = *(const v8bf*)(p);
  v8bf c1 = *(const v8bf*)(p + 8);
  v16bf r;
#pragma unroll
  for (int i = 0; i < 8; ++i) { r[i] = c0[i]; r[i + 8] = c1[i]; }
  return r;
}

// B fragment (32x16, bf16). Memory holds B^T row-major: Brow[n, k] with
// leading dim ld, so lane n reads 16 contiguous K values:
// lanes 0-15 hold K = k..k+15, lanes 16-31 hold K = k+16..k+31.
__device__ inline v16bf load_b_frag(const bf16* base, int col, int k, int ld,
                                    int lm, int grp) {
  return *(const v16bf*)(base + (size_t)(col + lm) * ld + k + grp * 16);
}

// ---------------------------------------------------------------- cvt
__global__ void __launch_bounds__(256)
cvt_f32_bf16(const float* __restrict__ src, bf16* __restrict__ dst, long n) {
  long i = ((long)blockIdx.x * blockDim.x + threadIdx.x) * 4;
  if (i + 3 < n) {
    float4 v = *(const float4*)(src + i);
    dst[i + 0] = (bf16)v.x; dst[i + 1] = (bf16)v.y;
    dst[i + 2] = (bf16)v.z; dst[i + 3] = (bf16)v.w;
  } else {
    for (long j = i; j < n; ++j) dst[j] = (bf16)src[j];
  }
}

// ---------------------------------------------------------------- bias
// bias[i,j] = 1 / log(relu(sum_t (|i-t|+eps)*wm_w[j,t] + wm_b[j]) + 2eps)
__global__ void __launch_bounds__(256)
bias_kernel(const float* __restrict__ wm_w, const float* __restrict__ wm_b,
            float* __restrict__ biasOut) {
  const int i = blockIdx.x;
  for (int j = threadIdx.x; j < SS; j += 256) {
    const float* wr = wm_w + (size_t)j * SS;
    float acc = 0.0f;
    for (int t = 0; t < SS; ++t) {
      float d = fabsf((float)(i - t)) + 1e-7f;
      acc = fmaf(d, wr[t], acc);
    }
    acc = fmaxf(acc + wm_b[j], 0.0f);
    biasOut[(size_t)i * SS + j] = 1.0f / logf(acc + 2.0e-7f);
  }
}

// ---------------------------------------------------------------- QKV GEMM
// Out[b,s,f] = sum_e X[b,s,e] * W[f,e].
// 8 waves arranged 4(M) x 2(N); wave tile 64x64 (16 f32 accumulators,
// 32 FLOP/byte of fragment traffic); block tile 256(M) x 128(N).
// transposed=1 stores Out as [E][S] (for V).
__global__ void __launch_bounds__(256)
gemm_xw_bf16(const bf16* __restrict__ X, const bf16* __restrict__ W,
             bf16* __restrict__ Out, int transposed) {
  const int b    = blockIdx.z;
  const int wv_  = threadIdx.x >> 5;
  const int lane = threadIdx.x & 31;
  const int lm   = lane & 15;
  const int grp  = lane >> 4;
  const int wm   = blockIdx.y * 256 + (wv_ & 3) * 64;
  const int wn   = blockIdx.x * 128 + (wv_ >> 2) * 64;
  const bf16* Xb = X + (size_t)b * SS * EE;

  v8f acc[4][4];
#pragma unroll
  for (int i = 0; i < 4; ++i)
#pragma unroll
    for (int j = 0; j < 4; ++j) acc[i][j] = v8f_zero();

  for (int k = 0; k < EE; k += 32) {
    v16bf af[4], bfr[4];
#pragma unroll
    for (int mt = 0; mt < 4; ++mt)
      af[mt] = load_a_frag(Xb, wm + mt * 16, k, EE, lm, grp);
#pragma unroll
    for (int nt = 0; nt < 4; ++nt)
      bfr[nt] = load_b_frag(W, wn + nt * 16, k, EE, lm, grp);
#pragma unroll
    for (int mt = 0; mt < 4; ++mt)
#pragma unroll
      for (int nt = 0; nt < 4; ++nt)
        acc[mt][nt] = wmma_bf16(af[mt], bfr[nt], acc[mt][nt]);
  }

  if (!transposed) {
    bf16* Ob = Out + (size_t)b * SS * EE;
#pragma unroll
    for (int mt = 0; mt < 4; ++mt)
#pragma unroll
      for (int nt = 0; nt < 4; ++nt)
#pragma unroll
        for (int v2 = 0; v2 < 8; ++v2) {
          int r = wm + mt * 16 + grp * 8 + v2;
          int c = wn + nt * 16 + lm;
          Ob[(size_t)r * EE + c] = (bf16)acc[mt][nt][v2];
        }
  } else {
    bf16* Ob = Out + (size_t)b * EE * SS;  // [E][S]
#pragma unroll
    for (int mt = 0; mt < 4; ++mt)
#pragma unroll
      for (int nt = 0; nt < 4; ++nt)
#pragma unroll
        for (int v2 = 0; v2 < 8; ++v2) {
          int r = wm + mt * 16 + grp * 8 + v2;     // t
          int c = wn + nt * 16 + lm;               // f
          Ob[(size_t)c * SS + r] = (bf16)acc[mt][nt][v2];
        }
  }
}

// ---------------------------------------------------------------- attention
// One block = (batch b, 32 query rows).  Full 32x1024 score tile in LDS.
// K-outer with 4 key/feature tiles held in accumulators (two passes):
// Q / P fragments are loaded 2x total instead of 8x.
__global__ void __launch_bounds__(256)
attn_kernel(const bf16* __restrict__ Q, const bf16* __restrict__ K,
            const bf16* __restrict__ Vt, const float* __restrict__ bias,
            const int* __restrict__ mask, const float* __restrict__ lin_w,
            const float* __restrict__ lin_b,
            float* __restrict__ OutF, float* __restrict__ Aw) {
  __shared__ float weiF[32 * SS];   // 128 KB (WGP has 320 KB LDS)
  __shared__ float awLds[32];
  bf16* weiB = (bf16*)weiF;         // bf16 row r occupies first half of f32 row r
                                    // -> bf16 leading dim = 2*SS elements
  const int b    = blockIdx.y;
  const int sB   = blockIdx.x * 32;
  const int wv_  = threadIdx.x >> 5;
  const int lane = threadIdx.x & 31;
  const int lm   = lane & 15;
  const int grp  = lane >> 4;

  const bf16* Qb = Q  + (size_t)b * SS * EE;
  const bf16* Kb = K  + (size_t)b * SS * EE;
  const bf16* Vb = Vt + (size_t)b * EE * SS;

  if (threadIdx.x < 32) awLds[threadIdx.x] = 0.0f;

  // ---- Phase 1: scores = Q K^T + bias, masked, into LDS (f32)
  for (int pass = 0; pass < 2; ++pass) {
    v8f acc0[4], acc1[4];
#pragma unroll
    for (int i = 0; i < 4; ++i) { acc0[i] = v8f_zero(); acc1[i] = v8f_zero(); }

    for (int k = 0; k < EE; k += 32) {
      v16bf a0 = load_a_frag(Qb, sB,      k, EE, lm, grp);
      v16bf a1 = load_a_frag(Qb, sB + 16, k, EE, lm, grp);
#pragma unroll
      for (int i = 0; i < 4; ++i) {
        const int tB = (wv_ + 8 * (pass * 4 + i)) * 16;
        v16bf bf = load_b_frag(Kb, tB, k, EE, lm, grp);
        acc0[i] = wmma_bf16(a0, bf, acc0[i]);
        acc1[i] = wmma_bf16(a1, bf, acc1[i]);
      }
    }
#pragma unroll
    for (int i = 0; i < 4; ++i) {
      const int t  = (wv_ + 8 * (pass * 4 + i)) * 16 + lm;
      const int mk = mask[b * SS + t];
#pragma unroll
      for (int v2 = 0; v2 < 8; ++v2) {
        int m0 = grp * 8 + v2;
        int m1 = 16 + m0;
        float s0 = acc0[i][v2] + bias[(size_t)(sB + m0) * SS + t];
        float s1 = acc1[i][v2] + bias[(size_t)(sB + m1) * SS + t];
        weiF[m0 * SS + t] = mk ? s0 : NEGV;
        weiF[m1 * SS + t] = mk ? s1 : NEGV;
      }
    }
  }
  __syncthreads();

  // ---- Phase 2: per-row softmax, re-quantize to bf16 in place
  for (int rr = 0; rr < 4; ++rr) {
    const int r = wv_ * 4 + rr;          // wave exclusively owns its rows
    float vals[32];
    float mx = -3.0e38f;
#pragma unroll
    for (int i = 0; i < 32; ++i) {
      vals[i] = weiF[r * SS + lane + 32 * i];
      mx = fmaxf(mx, vals[i]);
    }
#pragma unroll
    for (int off = 16; off > 0; off >>= 1) mx = fmaxf(mx, __shfl_xor(mx, off, 32));
    float sum = 0.0f;
#pragma unroll
    for (int i = 0; i < 32; ++i) { vals[i] = __expf(vals[i] - mx); sum += vals[i]; }
#pragma unroll
    for (int off = 16; off > 0; off >>= 1) sum += __shfl_xor(sum, off, 32);
    const float inv = 1.0f / sum;
#pragma unroll
    for (int i = 0; i < 32; ++i)
      weiB[(size_t)r * (2 * SS) + lane + 32 * i] = (bf16)(vals[i] * inv);
  }
  __syncthreads();

  // ---- Phase 3: out = P @ V  (A from LDS, B = V^T from global), + aw logits
  float awloc0[8], awloc1[8];
#pragma unroll
  for (int j = 0; j < 8; ++j) { awloc0[j] = 0.0f; awloc1[j] = 0.0f; }
  float* Ob = OutF + (size_t)b * SS * EE;
  const bf16* weiBc = (const bf16*)weiB;

  for (int pass = 0; pass < 2; ++pass) {
    v8f acc0[4], acc1[4];
#pragma unroll
    for (int i = 0; i < 4; ++i) { acc0[i] = v8f_zero(); acc1[i] = v8f_zero(); }

    for (int k = 0; k < SS; k += 32) {
      v16bf a0 = load_a_frag(weiBc, 0,  k, 2 * SS, lm, grp);
      v16bf a1 = load_a_frag(weiBc, 16, k, 2 * SS, lm, grp);
#pragma unroll
      for (int i = 0; i < 4; ++i) {
        const int fB = (wv_ + 8 * (pass * 4 + i)) * 16;
        v16bf bf = load_b_frag(Vb, fB, k, SS, lm, grp);
        acc0[i] = wmma_bf16(a0, bf, acc0[i]);
        acc1[i] = wmma_bf16(a1, bf, acc1[i]);
      }
    }
#pragma unroll
    for (int i = 0; i < 4; ++i) {
      const int f  = (wv_ + 8 * (pass * 4 + i)) * 16 + lm;
      const float lw = lin_w[f];
#pragma unroll
      for (int v2 = 0; v2 < 8; ++v2) {
        int m0 = grp * 8 + v2;
        int m1 = 16 + m0;
        Ob[(size_t)(sB + m0) * EE + f] = acc0[i][v2];
        Ob[(size_t)(sB + m1) * EE + f] = acc1[i][v2];
        awloc0[v2] = fmaf(acc0[i][v2], lw, awloc0[v2]);
        awloc1[v2] = fmaf(acc1[i][v2], lw, awloc1[v2]);
      }
    }
  }
#pragma unroll
  for (int v2 = 0; v2 < 8; ++v2) {
    atomicAdd(&awLds[grp * 8 + v2],      awloc0[v2]);
    atomicAdd(&awLds[16 + grp * 8 + v2], awloc1[v2]);
  }
  __syncthreads();
  if (threadIdx.x < 32)
    Aw[b * SS + sB + threadIdx.x] = awLds[threadIdx.x] + lin_b[0];
}

// ---------------------------------------------------------------- pooling
__global__ void __launch_bounds__(256)
pool_kernel(const float* __restrict__ OutF, const float* __restrict__ Aw,
            const int* __restrict__ mask, float* __restrict__ out) {
  __shared__ float w[SS];
  __shared__ float red[256];
  const int b = blockIdx.x;
  const int tid = threadIdx.x;

  float vloc[4];
  float mx = -3.0e38f;
#pragma unroll
  for (int i = 0; i < 4; ++i) {
    int t = tid + 256 * i;
    float v = mask[b * SS + t] ? Aw[b * SS + t] : NEGV;
    vloc[i] = v;
    mx = fmaxf(mx, v);
  }
  red[tid] = mx;
  __syncthreads();
  for (int s = 128; s > 0; s >>= 1) {
    if (tid < s) red[tid] = fmaxf(red[tid], red[tid + s]);
    __syncthreads();
  }
  const float bm = red[0];
  __syncthreads();

  float sum = 0.0f;
#pragma unroll
  for (int i = 0; i < 4; ++i) { vloc[i] = __expf(vloc[i] - bm); sum += vloc[i]; }
  red[tid] = sum;
  __syncthreads();
  for (int s = 128; s > 0; s >>= 1) {
    if (tid < s) red[tid] += red[tid + s];
    __syncthreads();
  }
  const float inv = 1.0f / red[0];
  __syncthreads();
#pragma unroll
  for (int i = 0; i < 4; ++i) w[tid + 256 * i] = vloc[i] * inv;
  __syncthreads();

  const float* Ob = OutF + (size_t)b * SS * EE;
#pragma unroll
  for (int j = 0; j < 4; ++j) {
    const int f = tid + 256 * j;
    float acc = 0.0f;
#pragma unroll 4
    for (int t = 0; t < SS; ++t) acc = fmaf(w[t], Ob[(size_t)t * EE + f], acc);
    out[(size_t)b * EE + f] = acc;
  }
}

// ---------------------------------------------------------------- launch
extern "C" void kernel_launch(void* const* d_in, const int* in_sizes, int n_in,
                              void* d_out, int out_size, void* d_ws, size_t ws_size,
                              hipStream_t stream) {
  const float* x     = (const float*)d_in[0];
  const int*   mask  = (const int*)  d_in[1];
  const float* wq    = (const float*)d_in[2];
  const float* wk    = (const float*)d_in[3];
  const float* wv    = (const float*)d_in[4];
  const float* wm_w  = (const float*)d_in[5];
  const float* wm_b  = (const float*)d_in[6];
  const float* lin_w = (const float*)d_in[7];
  const float* lin_b = (const float*)d_in[8];
  (void)in_sizes; (void)n_in; (void)out_size; (void)ws_size;

  char* ws = (char*)d_ws;
  size_t off = 0;
  auto take = [&](size_t bytes) {
    char* p = ws + off;
    off += (bytes + 255) & ~(size_t)255;
    return p;
  };
  bf16*  xb    = (bf16*) take((size_t)BB * SS * EE * 2);
  bf16*  wqb   = (bf16*) take((size_t)EE * EE * 2);
  bf16*  wkb   = (bf16*) take((size_t)EE * EE * 2);
  bf16*  wvb   = (bf16*) take((size_t)EE * EE * 2);
  bf16*  Qb    = (bf16*) take((size_t)BB * SS * EE * 2);
  bf16*  Kb    = (bf16*) take((size_t)BB * SS * EE * 2);
  bf16*  Vt    = (bf16*) take((size_t)BB * SS * EE * 2);   // [B][E][S]
  float* biasB = (float*)take((size_t)SS * SS * 4);
  float* OutF  = (float*)take((size_t)BB * SS * EE * 4);
  float* Aw    = (float*)take((size_t)BB * SS * 4);

  const long nx = (long)BB * SS * EE;
  const long nw = (long)EE * EE;
  cvt_f32_bf16<<<dim3((unsigned)(nx / 1024)), dim3(256), 0, stream>>>(x,  xb,  nx);
  cvt_f32_bf16<<<dim3((unsigned)(nw / 1024)), dim3(256), 0, stream>>>(wq, wqb, nw);
  cvt_f32_bf16<<<dim3((unsigned)(nw / 1024)), dim3(256), 0, stream>>>(wk, wkb, nw);
  cvt_f32_bf16<<<dim3((unsigned)(nw / 1024)), dim3(256), 0, stream>>>(wv, wvb, nw);

  bias_kernel<<<dim3(SS), dim3(256), 0, stream>>>(wm_w, wm_b, biasB);

  dim3 gg(EE / 128, SS / 256, BB);
  gemm_xw_bf16<<<gg, dim3(256), 0, stream>>>(xb, wqb, Qb, 0);
  gemm_xw_bf16<<<gg, dim3(256), 0, stream>>>(xb, wkb, Kb, 0);
  gemm_xw_bf16<<<gg, dim3(256), 0, stream>>>(xb, wvb, Vt, 1);

  attn_kernel<<<dim3(SS / 32, BB), dim3(256), 0, stream>>>(
      Qb, Kb, Vt, biasB, mask, lin_w, lin_b, OutF, Aw);

  pool_kernel<<<dim3(BB), dim3(256), 0, stream>>>(OutF, Aw, mask, (float*)d_out);
}
